// CondAttLSTM_56530359550210
// MI455X (gfx1250) — matrix operand order repaired and checked by reading, not compiled, not run on hardware
//
#include <hip/hip_runtime.h>

// ---------------------------------------------------------------------------
// Problem constants
// ---------------------------------------------------------------------------
#define BB   32
#define TT   128
#define LL   64
#define DIN  512
#define HH   512
#define DCTX 512
#define ATT  256
#define NGATE 2048   // 4*H, gate-interleaved: n = 64*q + 16*g + c, hcol = 16*q + c

typedef __attribute__((ext_vector_type(16))) __bf16          v16bf;
typedef __attribute__((ext_vector_type(16))) unsigned short  v16us;
typedef __attribute__((ext_vector_type(8)))  unsigned short  v8us;
typedef __attribute__((ext_vector_type(8)))  float           v8f;

union Frag16 { v16bf v; v16us u; v8us h[2]; };

// ---------------------------------------------------------------------------
// bf16 helpers (bit-level, round-to-nearest-even)
// ---------------------------------------------------------------------------
__device__ inline unsigned short f2bf(float x) {
    union { float f; unsigned int u; } v; v.f = x;
    unsigned int u = v.u;
    unsigned int r = (u + 0x7FFFu + ((u >> 16) & 1u)) >> 16;
    return (unsigned short)r;
}
__device__ inline float bf2f(unsigned short b) {
    union { unsigned int u; float f; } v; v.u = ((unsigned int)b) << 16;
    return v.f;
}
__device__ inline float sigmoidf(float x) { return 1.0f / (1.0f + __expf(-x)); }

// ---------------------------------------------------------------------------
// WMMA fragment loaders (wave32, bf16 16x16x32; this code's contiguous k order, same for A and B)
// A 16x32: lanes 0-15 hold row M=lane, K=0..15;
//          lanes 16-31 hold row M=lane-16, K=16..31.
// B 32x16 (stored N-major [N][K]): lane n (0-15) K=0..15, lane n+16 K=16..31.
// ---------------------------------------------------------------------------
__device__ inline v16bf load_a_frag(const unsigned short* __restrict__ A,
                                    int ldk, int row0, int ko) {
    int lane = threadIdx.x & 31;
    int r  = row0 + (lane & 15);
    int kb = ko + ((lane & 16) ? 16 : 0);
    const unsigned short* p = A + (size_t)r * ldk + kb;
    Frag16 f;
    f.h[0] = *(const v8us*)(p);        // K = kb .. kb+7
    f.h[1] = *(const v8us*)(p + 8);    // K = kb+8 .. kb+15
    return f.v;
}
__device__ inline v16bf load_b_frag(const unsigned short* __restrict__ Bt,
                                    int ldk, int n0, int ko) {
    int lane = threadIdx.x & 31;
    int n = n0 + (lane & 15);
    int k = ko + ((lane & 16) ? 16 : 0);
    Frag16 f;
    f.u = *(const v16us*)(Bt + (size_t)n * ldk + k);
    return f.v;
}
__device__ inline v8f wmma_bf16(v16bf a, v16bf b, v8f c) {
    return __builtin_amdgcn_wmma_f32_16x16x32_bf16(
        /*neg_a=*/false, a, /*neg_b=*/false, b,
        /*c_mod=*/(short)0, c, /*reuse_a=*/false, /*reuse_b=*/false);
}

// ---------------------------------------------------------------------------
// Packing kernels (one-time): f32 -> bf16, transposed to N-major, gate-interleaved
// ---------------------------------------------------------------------------
__global__ void pack_W_kernel(const float* __restrict__ Wi, const float* __restrict__ Wf,
                              const float* __restrict__ Wc, const float* __restrict__ Wo,
                              const float* __restrict__ bi, const float* __restrict__ bff,
                              const float* __restrict__ bc, const float* __restrict__ bo,
                              unsigned short* __restrict__ Wt /*[2048][512]*/,
                              float* __restrict__ biascat /*[2048]*/) {
    int n = blockIdx.x;
    int g = (n >> 4) & 3;
    int h = ((n >> 6) << 4) + (n & 15);
    const float* W = (g == 0) ? Wi : (g == 1) ? Wf : (g == 2) ? Wc : Wo;
    const float* bb = (g == 0) ? bi : (g == 1) ? bff : (g == 2) ? bc : bo;
    for (int k = threadIdx.x; k < DIN; k += blockDim.x)
        Wt[(size_t)n * DIN + k] = f2bf(W[(size_t)k * HH + h]);
    if (threadIdx.x == 0) biascat[n] = bb[h];
}

__global__ void pack_UC_kernel(const float* __restrict__ Ui, const float* __restrict__ Uf,
                               const float* __restrict__ Uc, const float* __restrict__ Uo,
                               const float* __restrict__ Ci, const float* __restrict__ Cf,
                               const float* __restrict__ Cc, const float* __restrict__ Co,
                               unsigned short* __restrict__ UCt /*[2048][1024]*/) {
    int n = blockIdx.x;
    int g = (n >> 4) & 3;
    int h = ((n >> 6) << 4) + (n & 15);
    const float* U = (g == 0) ? Ui : (g == 1) ? Uf : (g == 2) ? Uc : Uo;
    const float* C = (g == 0) ? Ci : (g == 1) ? Cf : (g == 2) ? Cc : Co;
    for (int k = threadIdx.x; k < 2 * HH; k += blockDim.x) {
        float v = (k < HH) ? U[(size_t)k * HH + h] : C[(size_t)(k - HH) * HH + h];
        UCt[(size_t)n * (2 * HH) + k] = f2bf(v);
    }
}

__global__ void pack_att_kernel(const float* __restrict__ attHW1 /*[512][256]*/,
                                const float* __restrict__ attCtxW1 /*[512][256]*/,
                                unsigned short* __restrict__ AhT /*[256][512]*/,
                                unsigned short* __restrict__ AcT /*[256][512]*/) {
    int a = blockIdx.x;
    for (int k = threadIdx.x; k < DCTX; k += blockDim.x) {
        AhT[(size_t)a * DCTX + k] = f2bf(attHW1[(size_t)k * ATT + a]);
        AcT[(size_t)a * DCTX + k] = f2bf(attCtxW1[(size_t)k * ATT + a]);
    }
}

__global__ void cvt_bf_kernel(const float* __restrict__ src,
                              unsigned short* __restrict__ dst, int n) {
    for (int i = blockIdx.x * blockDim.x + threadIdx.x; i < n; i += gridDim.x * blockDim.x)
        dst[i] = f2bf(src[i]);
}

__global__ void zero_kernel(float* __restrict__ p, int n) {
    for (int i = blockIdx.x * blockDim.x + threadIdx.x; i < n; i += gridDim.x * blockDim.x)
        p[i] = 0.0f;
}

// ---------------------------------------------------------------------------
// bf16 WMMA GEMM: C[M][N] = A[M][K] * Bt[N][K]^T + bias[N]
// Block = 256 threads = 8 waves (2x4), block tile 64(M) x 128(N), wave tile 32x32.
// If Tdim > 0: output rows remapped m=(b*Tdim+t) -> (t*(M/Tdim)+b)  (time-major).
// ---------------------------------------------------------------------------
__global__ void gemm_bf16_kernel(const unsigned short* __restrict__ A,
                                 const unsigned short* __restrict__ Bt,
                                 const float* __restrict__ bias,
                                 float* __restrict__ C,
                                 int M, int N, int K, int Tdim) {
    int wave = threadIdx.x >> 5;
    int lane = threadIdx.x & 31;
    int mw = wave >> 2, nw = wave & 3;
    int m0 = blockIdx.y * 64 + mw * 32;
    int n0 = blockIdx.x * 128 + nw * 32;

    v8f acc00 = {}, acc01 = {}, acc10 = {}, acc11 = {};
    for (int ko = 0; ko < K; ko += 32) {
        v16bf a0 = load_a_frag(A, K, m0,      ko);
        v16bf a1 = load_a_frag(A, K, m0 + 16, ko);
        v16bf b0 = load_b_frag(Bt, K, n0,      ko);
        v16bf b1 = load_b_frag(Bt, K, n0 + 16, ko);
        acc00 = wmma_bf16(a0, b0, acc00);
        acc01 = wmma_bf16(a0, b1, acc01);
        acc10 = wmma_bf16(a1, b0, acc10);
        acc11 = wmma_bf16(a1, b1, acc11);
    }

    int nn = lane & 15;
    int mb = (lane & 16) ? 8 : 0;
    v8f* accs[2][2] = { {&acc00, &acc01}, {&acc10, &acc11} };
    for (int mt = 0; mt < 2; ++mt) {
        for (int nt = 0; nt < 2; ++nt) {
            int n = n0 + nt * 16 + nn;
            float bv = bias ? bias[n] : 0.0f;
            for (int v = 0; v < 8; ++v) {
                int m = m0 + mt * 16 + mb + v;
                int orow = m;
                if (Tdim > 0) {
                    int bidx = m / Tdim, tidx = m % Tdim;
                    orow = tidx * (M / Tdim) + bidx;
                }
                C[(size_t)orow * N + n] = (*accs[mt][nt])[v] + bv;
            }
        }
    }
}

// ---------------------------------------------------------------------------
// Per-step attention: one block per batch row (32 blocks x 256 threads).
// Produces softmax-weighted ctx_vec, writes ctxs output and bf16 [h|ctx] row.
// ---------------------------------------------------------------------------
__global__ void attention_kernel(const float* __restrict__ hstate /*[32][512]*/,
                                 const unsigned short* __restrict__ AhT /*[256][512]*/,
                                 const float* __restrict__ ctrans /*[B*L][256]*/,
                                 const float* __restrict__ attW2 /*[256]*/,
                                 const float* __restrict__ attB2 /*[1]*/,
                                 const float* __restrict__ context /*[B][L][512]*/,
                                 const int* __restrict__ cmask /*[B][L]*/,
                                 unsigned short* __restrict__ hcat /*[32][1024]*/,
                                 float* __restrict__ ctxs_out /*[B][T][512]*/,
                                 int t) {
    __shared__ float h_sh[HH];
    __shared__ float s_sh[ATT];
    __shared__ float red[256];
    __shared__ float a_sh[LL];
    __shared__ float invsum;

    int b = blockIdx.x, tid = threadIdx.x;
    h_sh[tid]       = hstate[b * HH + tid];
    h_sh[tid + 256] = hstate[b * HH + 256 + tid];
    __syncthreads();

    // s[a] = h . att_h_W1[:,a]
    {
        const unsigned short* row = AhT + (size_t)tid * HH;
        float acc = 0.0f;
        #pragma unroll 8
        for (int k = 0; k < HH; ++k) acc += h_sh[k] * bf2f(row[k]);
        s_sh[tid] = acc;
    }
    __syncthreads();

    // att_raw[l] = sum_a tanh(ctx_trans[b,l,a] + s[a]) * W2[a]   (4 partials per l)
    int l = tid & 63, part = tid >> 6;
    {
        const float* ct = ctrans + ((size_t)b * LL + l) * ATT;
        float acc = 0.0f;
        for (int a = part * 64; a < part * 64 + 64; ++a)
            acc += tanhf(ct[a] + s_sh[a]) * attW2[a];
        red[tid] = acc;
    }
    __syncthreads();
    if (tid < 64) {
        float r = red[l] + red[l + 64] + red[l + 128] + red[l + 192] + attB2[0];
        a_sh[l] = __expf(r) * (float)cmask[b * LL + l];   // raw exp, as reference
    }
    __syncthreads();
    if (tid == 0) {
        float s = 0.0f;
        for (int i = 0; i < LL; ++i) s += a_sh[i];
        invsum = 1.0f / s;
    }
    __syncthreads();
    if (tid < 64) a_sh[tid] *= invsum;
    __syncthreads();

    // ctx_vec[d] = sum_l a[l] * context[b,l,d];  emit ctxs + bf16 [h|ctx] row
    for (int d = tid; d < DCTX; d += 256) {
        const float* cb = context + (size_t)b * LL * DCTX + d;
        float cv = 0.0f;
        #pragma unroll 8
        for (int ll2 = 0; ll2 < LL; ++ll2) cv += a_sh[ll2] * cb[(size_t)ll2 * DCTX];
        ctxs_out[((size_t)b * TT + t) * DCTX + d] = cv;
        hcat[b * (2 * HH) + HH + d] = f2bf(cv);
        hcat[b * (2 * HH) + d]      = f2bf(h_sh[d]);
    }
}

// ---------------------------------------------------------------------------
// Per-step fused gate GEMM + LSTM update.
// gates = xg[t] + [h|ctx] (32x1024 bf16) * UCt^T (1024x2048 bf16), f32 accum.
// Grid: 16 blocks x 256 thr; block covers 2 q-groups (N=128). Wave (ql,g) owns
// the 32x16 tile of gate g for q-group ql; tiles staged in LDS, then pointwise.
// ---------------------------------------------------------------------------
__global__ void gate_kernel(const unsigned short* __restrict__ hcat /*[32][1024]*/,
                            const unsigned short* __restrict__ UCt  /*[2048][1024]*/,
                            const float* __restrict__ xg   /*[T][32][2048] time-major*/,
                            const float* __restrict__ mask /*[B][T]*/,
                            float* __restrict__ hstate, float* __restrict__ cstate,
                            float* __restrict__ hs_out, float* __restrict__ cs_out,
                            int t) {
    __shared__ float tiles[2][4][BB][16];

    int tid = threadIdx.x;
    int wave = tid >> 5, lane = tid & 31;
    int ql = wave >> 2, g = wave & 3;
    int q  = blockIdx.x * 2 + ql;
    int n0 = q * 64 + g * 16;

    v8f acc0 = {}, acc1 = {};
    for (int ko = 0; ko < 2 * HH; ko += 32) {
        v16bf a0 = load_a_frag(hcat, 2 * HH, 0,  ko);
        v16bf a1 = load_a_frag(hcat, 2 * HH, 16, ko);
        v16bf bf = load_b_frag(UCt,  2 * HH, n0, ko);
        acc0 = wmma_bf16(a0, bf, acc0);
        acc1 = wmma_bf16(a1, bf, acc1);
    }

    // stage pre-activations (+ precomputed x-projection) into LDS
    int nn = lane & 15;
    int mb = (lane & 16) ? 8 : 0;
    const float* xgrow = xg + (size_t)t * BB * NGATE;   // contiguous 256KB slab for step t
    #pragma unroll
    for (int v = 0; v < 8; ++v) {
        int m0 = mb + v, m1 = 16 + mb + v;
        tiles[ql][g][m0][nn] = acc0[v] + xgrow[(size_t)m0 * NGATE + n0 + nn];
        tiles[ql][g][m1][nn] = acc1[v] + xgrow[(size_t)m1 * NGATE + n0 + nn];
    }
    __syncthreads();

    // pointwise LSTM update: i/f/c/o for same (b, hcol) are in this block's LDS, 512 floats apart
    for (int e = tid; e < 2 * BB * 16; e += 256) {
        int ql2 = e >> 9;
        int rem = e & 511;
        int m = rem >> 4, c = rem & 15;
        int qq = blockIdx.x * 2 + ql2;
        int hcol = qq * 16 + c;
        int b = m;
        float iv = sigmoidf(tiles[ql2][0][m][c]);
        float fv = sigmoidf(tiles[ql2][1][m][c]);
        float cc = tanhf   (tiles[ql2][2][m][c]);
        float ov = sigmoidf(tiles[ql2][3][m][c]);
        float c_old = cstate[b * HH + hcol];
        float h_old = hstate[b * HH + hcol];
        float c_new = fv * c_old + iv * cc;
        float h_new = ov * tanhf(c_new);
        float mt = mask[b * TT + t];
        h_new = (1.0f - mt) * h_old + mt * h_new;
        c_new = (1.0f - mt) * c_old + mt * c_new;
        hstate[b * HH + hcol] = h_new;
        cstate[b * HH + hcol] = c_new;
        hs_out[((size_t)b * TT + t) * HH + hcol] = h_new;
        cs_out[((size_t)b * TT + t) * HH + hcol] = c_new;
    }
}

// ---------------------------------------------------------------------------
// Host launcher
// ---------------------------------------------------------------------------
extern "C" void kernel_launch(void* const* d_in, const int* in_sizes, int n_in,
                              void* d_out, int out_size, void* d_ws, size_t ws_size,
                              hipStream_t stream) {
    (void)in_sizes; (void)n_in; (void)out_size; (void)ws_size;

    const float* X       = (const float*)d_in[0];
    const float* context = (const float*)d_in[1];
    const float* mask    = (const float*)d_in[2];
    const int*   cmask   = (const int*)d_in[3];
    const float* W[4]  = {(const float*)d_in[4],  (const float*)d_in[8],
                          (const float*)d_in[12], (const float*)d_in[16]};
    const float* U[4]  = {(const float*)d_in[5],  (const float*)d_in[9],
                          (const float*)d_in[13], (const float*)d_in[17]};
    const float* Cg[4] = {(const float*)d_in[6],  (const float*)d_in[10],
                          (const float*)d_in[14], (const float*)d_in[18]};
    const float* bg[4] = {(const float*)d_in[7],  (const float*)d_in[11],
                          (const float*)d_in[15], (const float*)d_in[19]};
    const float* attCtxW1 = (const float*)d_in[20];
    const float* attHW1   = (const float*)d_in[21];
    const float* attB1    = (const float*)d_in[22];
    const float* attW2    = (const float*)d_in[23];
    const float* attB2    = (const float*)d_in[24];

    char* ws = (char*)d_ws;
    size_t o = 0;
    auto take = [&](size_t bytes) {
        char* p = ws + o;
        o = (o + bytes + 255) & ~(size_t)255;
        return p;
    };
    unsigned short* Wt     = (unsigned short*)take((size_t)NGATE * DIN * 2);       // 2 MB
    unsigned short* UCt    = (unsigned short*)take((size_t)NGATE * 2 * HH * 2);    // 4 MB
    unsigned short* AhT    = (unsigned short*)take((size_t)ATT * HH * 2);
    unsigned short* AcT    = (unsigned short*)take((size_t)ATT * DCTX * 2);
    unsigned short* Xbf    = (unsigned short*)take((size_t)BB * TT * DIN * 2);     // 4 MB
    unsigned short* Ctxbf  = (unsigned short*)take((size_t)BB * LL * DCTX * 2);    // 2 MB
    float* biascat = (float*)take((size_t)NGATE * 4);
    float* xg      = (float*)take((size_t)BB * TT * NGATE * 4);                    // 32 MB
    float* ctrans  = (float*)take((size_t)BB * LL * ATT * 4);                      // 2 MB
    float* hstate  = (float*)take((size_t)BB * HH * 4);
    float* cstate  = (float*)take((size_t)BB * HH * 4);
    unsigned short* hcat = (unsigned short*)take((size_t)BB * 2 * HH * 2);

    float* hs_out   = (float*)d_out;
    float* cs_out   = hs_out + (size_t)BB * TT * HH;
    float* ctxs_out = cs_out + (size_t)BB * TT * HH;

    // one-time packing / precompute
    pack_W_kernel <<<NGATE, 256, 0, stream>>>(W[0], W[1], W[2], W[3],
                                              bg[0], bg[1], bg[2], bg[3], Wt, biascat);
    pack_UC_kernel<<<NGATE, 256, 0, stream>>>(U[0], U[1], U[2], U[3],
                                              Cg[0], Cg[1], Cg[2], Cg[3], UCt);
    pack_att_kernel<<<ATT, 256, 0, stream>>>(attHW1, attCtxW1, AhT, AcT);
    cvt_bf_kernel<<<2048, 256, 0, stream>>>(X,       Xbf,   BB * TT * DIN);
    cvt_bf_kernel<<<1024, 256, 0, stream>>>(context, Ctxbf, BB * LL * DCTX);
    zero_kernel<<<64, 256, 0, stream>>>(hstate, BB * HH);
    zero_kernel<<<64, 256, 0, stream>>>(cstate, BB * HH);

    // xg[t][b][:] = X[b,t,:] @ Wcat + biascat   (time-major output)
    gemm_bf16_kernel<<<dim3(NGATE / 128, (BB * TT) / 64), 256, 0, stream>>>(
        Xbf, Wt, biascat, xg, BB * TT, NGATE, DIN, TT);
    // ctx_trans = context @ att_ctx_W1 + att_b1
    gemm_bf16_kernel<<<dim3(ATT / 128, (BB * LL) / 64), 256, 0, stream>>>(
        Ctxbf, AcT, attB1, ctrans, BB * LL, ATT, DCTX, 0);

    // recurrence
    for (int t = 0; t < TT; ++t) {
        attention_kernel<<<BB, 256, 0, stream>>>(hstate, AhT, ctrans, attW2, attB2,
                                                 context, cmask, hcat, ctxs_out, t);
        gate_kernel<<<NGATE / 128, 256, 0, stream>>>(hcat, UCt, xg, mask,
                                                     hstate, cstate, hs_out, cs_out, t);
    }
}
